// StochasticGIN_2997887173238
// MI455X (gfx1250) — matrix operand order repaired and checked by reading, not compiled, not run on hardware
//
#include <hip/hip_runtime.h>
#include <hip/hip_bf16.h>

#define GN 50000
#define GE 800000
#define GD 128
#define GL 3
#define BN_EPS 1e-5f

typedef __attribute__((ext_vector_type(16))) _Float16 v16h;
typedef __attribute__((ext_vector_type(8)))  float    v8f;

// ---------------------------------------------------------------------------
// Pack W (f32 row-major [k][n], per layer 128x128) into WMMA B-operand layout:
// Bp[((kt*8+ct)*32+lane)*16 + j] = W[kt*32 + (lane>>4)*8 + (j&7) + (j>>3)*16][ct*16 + (lane&15)]
// so the GEMM does one contiguous 32-byte v16h load per B tile per lane.
// ---------------------------------------------------------------------------
__global__ void gin_pack_w(const float* __restrict__ W, _Float16* __restrict__ Bp, int total) {
    int idx = blockIdx.x * 256 + threadIdx.x;
    if (idx >= total) return;
    int layer = idx >> 14;           // D*D = 16384 per layer
    int t     = idx & 16383;
    int j    = t & 15;
    int lane = (t >> 4) & 31;
    int ct   = (t >> 9) & 7;
    int kt   = t >> 12;
    int n = ct * 16 + (lane & 15);
    int k = kt * 32 + ((lane >> 4) << 3) + (j & 7) + ((j >> 3) << 4);
    Bp[idx] = (_Float16)W[(size_t)layer * 16384 + k * GD + n];
}

// h -> (h_buf, r_buf)  (r starts as h; agg adds on top => r = h + agg)
__global__ void gin_copy2(const float* __restrict__ in, float* __restrict__ a,
                          float* __restrict__ b, int n) {
    int i = blockIdx.x * 256 + threadIdx.x;
    if (i < n) { float v = in[i]; a[i] = v; b[i] = v; }
}

__global__ void gin_zero(float* __restrict__ p, int n) {
    int i = blockIdx.x * 256 + threadIdx.x;
    if (i < n) p[i] = 0.f;
}

// ---------------------------------------------------------------------------
// Edge scatter: one wave32 per edge, 4 floats per lane (128 feats).
// r[dst] += w * h[src]  via global_atomic_add_f32 (L2-resident working set).
// ---------------------------------------------------------------------------
__global__ void gin_scatter(const float* __restrict__ h, const int* __restrict__ src,
                            const int* __restrict__ dst, const float* __restrict__ w,
                            float* __restrict__ r, int nedges) {
    int e = blockIdx.x * 8 + (threadIdx.x >> 5);
    if (e >= nedges) return;
    int lane = threadIdx.x & 31;
    int s = src[e], d = dst[e];
    float wt = w[e];
    const float4 v = *(const float4*)(h + (size_t)s * GD + lane * 4);
    float* p = r + (size_t)d * GD + lane * 4;
    atomicAdd(p + 0, wt * v.x);
    atomicAdd(p + 1, wt * v.y);
    atomicAdd(p + 2, wt * v.z);
    atomicAdd(p + 3, wt * v.w);
}

// ---------------------------------------------------------------------------
// WMMA GEMM: out[NxD] = act(A)[NxD] @ W[DxD] + bias, W pre-packed (Bp).
// One wave per 16-row stripe; 8 col tiles x 4 k-steps of v_wmma_f32_16x16x32_f16.
// APPLY_BN=1 fuses  a' = relu(a*scale[k]+shift[k])  into the A-operand load
// (compile-time: no per-element branches).  A/scale/shift loads are float4.
// ---------------------------------------------------------------------------
template <int APPLY_BN>
__global__ __launch_bounds__(256)
void gin_gemm_wmma(const float* __restrict__ A, const _Float16* __restrict__ Bp,
                   const float* __restrict__ bias, const float* __restrict__ scale,
                   const float* __restrict__ shift, float* __restrict__ out, int nrows) {
    const int lane = threadIdx.x & 31;
    const int wave = threadIdx.x >> 5;
    const int m0 = blockIdx.x * 128 + wave * 16;
    if (m0 >= nrows) return;                      // uniform per wave; EXEC stays all-ones
    const int arow = m0 + (lane & 15);
    const int kgrp = (lane >> 4) * 8;             // ISA 16-bit A layout lane-group offset
    const float4* __restrict__ arow4 = (const float4*)(A + (size_t)arow * GD);
    const float4* __restrict__ sc4   = (const float4*)scale;
    const float4* __restrict__ sh4   = (const float4*)shift;

    v8f acc[8] = {};
    #pragma unroll
    for (int kt = 0; kt < 4; ++kt) {
        // Lane's A slice for this k-step: K = kt*32+kgrp+[0..7] and +16+[0..7]
        const int f4b = kt * 8 + (kgrp >> 2);     // float4 index of first run
        float4 p0 = arow4[f4b + 0];
        float4 p1 = arow4[f4b + 1];
        float4 p2 = arow4[f4b + 4];               // +16 floats
        float4 p3 = arow4[f4b + 5];
        if (APPLY_BN) {
            float4 s0 = sc4[f4b + 0], s1 = sc4[f4b + 1], s2 = sc4[f4b + 4], s3 = sc4[f4b + 5];
            float4 t0 = sh4[f4b + 0], t1 = sh4[f4b + 1], t2 = sh4[f4b + 4], t3 = sh4[f4b + 5];
            p0.x = fmaxf(p0.x * s0.x + t0.x, 0.f); p0.y = fmaxf(p0.y * s0.y + t0.y, 0.f);
            p0.z = fmaxf(p0.z * s0.z + t0.z, 0.f); p0.w = fmaxf(p0.w * s0.w + t0.w, 0.f);
            p1.x = fmaxf(p1.x * s1.x + t1.x, 0.f); p1.y = fmaxf(p1.y * s1.y + t1.y, 0.f);
            p1.z = fmaxf(p1.z * s1.z + t1.z, 0.f); p1.w = fmaxf(p1.w * s1.w + t1.w, 0.f);
            p2.x = fmaxf(p2.x * s2.x + t2.x, 0.f); p2.y = fmaxf(p2.y * s2.y + t2.y, 0.f);
            p2.z = fmaxf(p2.z * s2.z + t2.z, 0.f); p2.w = fmaxf(p2.w * s2.w + t2.w, 0.f);
            p3.x = fmaxf(p3.x * s3.x + t3.x, 0.f); p3.y = fmaxf(p3.y * s3.y + t3.y, 0.f);
            p3.z = fmaxf(p3.z * s3.z + t3.z, 0.f); p3.w = fmaxf(p3.w * s3.w + t3.w, 0.f);
        }
        v16h a;
        a[0]  = (_Float16)p0.x; a[1]  = (_Float16)p0.y; a[2]  = (_Float16)p0.z; a[3]  = (_Float16)p0.w;
        a[4]  = (_Float16)p1.x; a[5]  = (_Float16)p1.y; a[6]  = (_Float16)p1.z; a[7]  = (_Float16)p1.w;
        a[8]  = (_Float16)p2.x; a[9]  = (_Float16)p2.y; a[10] = (_Float16)p2.z; a[11] = (_Float16)p2.w;
        a[12] = (_Float16)p3.x; a[13] = (_Float16)p3.y; a[14] = (_Float16)p3.z; a[15] = (_Float16)p3.w;

        #pragma unroll
        for (int ct = 0; ct < 8; ++ct) {
            v16h b = *(const v16h*)(Bp + ((((kt * 8 + ct) * 32) + lane) << 4));
            acc[ct] = __builtin_amdgcn_wmma_f32_16x16x32_f16(
                false, a, false, b, (short)0, acc[ct], false, false);
        }
    }
    // C layout: VGPR v -> row m0 + v + 8*(lane>>4), col = ct*16 + (lane&15)
    const int rbase = m0 + ((lane >> 4) << 3);
    const int cmod  = lane & 15;
    #pragma unroll
    for (int ct = 0; ct < 8; ++ct) {
        const float bc = bias[ct * 16 + cmod];
        float* op = out + (size_t)rbase * GD + ct * 16 + cmod;
        #pragma unroll
        for (int v = 0; v < 8; ++v)
            op[(size_t)v * GD] = acc[ct][v] + bc;
    }
}

// Per-column sum / sum-of-squares (coalesced: 128 threads = 128 columns).
__global__ void gin_colstats(const float* __restrict__ x, float* __restrict__ stats, int nrows) {
    int col = threadIdx.x;
    int r0 = blockIdx.x * 256;
    int r1 = r0 + 256; if (r1 > nrows) r1 = nrows;
    float s = 0.f, q = 0.f;
    for (int r = r0; r < r1; ++r) {
        float v = x[(size_t)r * GD + col];
        s += v; q += v * v;
    }
    atomicAdd(&stats[col], s);
    atomicAdd(&stats[GD + col], q);
}

// stats -> affine BN params: scale = g*rsqrt(var+eps), shift = b - mean*scale
__global__ void gin_bnparams(const float* __restrict__ stats, const float* __restrict__ g,
                             const float* __restrict__ b, float* __restrict__ scale,
                             float* __restrict__ shift, float invN) {
    int c = threadIdx.x;
    float m   = stats[c] * invN;
    float var = stats[GD + c] * invN - m * m;
    float s = g[c] * rsqrtf(var + BN_EPS);
    scale[c] = s;
    shift[c] = b[c] - m * s;
}

// y = relu(x*scale+shift) in place, fused stats accumulation for the NEXT BN.
__global__ void gin_apply_stats(float* __restrict__ x, const float* __restrict__ scale,
                                const float* __restrict__ shift, float* __restrict__ stats,
                                int nrows) {
    int col = threadIdx.x;
    int r0 = blockIdx.x * 256;
    int r1 = r0 + 256; if (r1 > nrows) r1 = nrows;
    float sc = scale[col], sh = shift[col];
    float s = 0.f, q = 0.f;
    for (int r = r0; r < r1; ++r) {
        size_t idx = (size_t)r * GD + col;
        float v = fmaxf(x[idx] * sc + sh, 0.f);
        x[idx] = v;
        s += v; q += v * v;
    }
    atomicAdd(&stats[col], s);
    atomicAdd(&stats[GD + col], q);
}

// Final outer BN+ReLU: write next-layer h and re-init r (= h) simultaneously.
__global__ void gin_final(const float* __restrict__ y, const float* __restrict__ scale,
                          const float* __restrict__ shift, float* __restrict__ hO,
                          float* __restrict__ rO, int n) {
    int i = blockIdx.x * 256 + threadIdx.x;
    if (i < n) {
        int col = i & (GD - 1);
        float v = fmaxf(y[i] * scale[col] + shift[col], 0.f);
        hO[i] = v;
        rO[i] = v;
    }
}

extern "C" void kernel_launch(void* const* d_in, const int* in_sizes, int n_in,
                              void* d_out, int out_size, void* d_ws, size_t ws_size,
                              hipStream_t stream) {
    const float* h_in  = (const float*)d_in[0];
    const int*   srcA  = (const int*)d_in[1];
    const int*   dstA  = (const int*)d_in[2];
    const float* ewA   = (const float*)d_in[3];
    const float* W1    = (const float*)d_in[4];
    const float* b1    = (const float*)d_in[5];
    const float* bn1_g = (const float*)d_in[6];
    const float* bn1_b = (const float*)d_in[7];
    const float* W2    = (const float*)d_in[8];
    const float* b2    = (const float*)d_in[9];
    const float* bna_g = (const float*)d_in[10];
    const float* bna_b = (const float*)d_in[11];
    const float* bno_g = (const float*)d_in[12];
    const float* bno_b = (const float*)d_in[13];

    const size_t ND = (size_t)GN * GD;
    char* ws = (char*)d_ws;
    size_t off = 0;
    auto carve = [&](size_t bytes) -> char* {
        char* p = ws + off;
        off = (off + bytes + 255) & ~(size_t)255;
        return p;
    };
    float*    h_buf  = (float*)carve(ND * 4);
    float*    r_buf  = (float*)carve(ND * 4);          // also GEMM2 output (x2)
    float*    x_buf  = (float*)carve(ND * 4);
    _Float16* Bp1    = (_Float16*)carve((size_t)GL * GD * GD * 2);
    _Float16* Bp2    = (_Float16*)carve((size_t)GL * GD * GD * 2);
    float*    stats  = (float*)carve(2 * GD * 4);
    float*    sA     = (float*)carve(GD * 4);
    float*    tA     = (float*)carve(GD * 4);
    float*    sB     = (float*)carve(GD * 4);
    float*    tB     = (float*)carve(GD * 4);
    float*    sC     = (float*)carve(GD * 4);
    float*    tC     = (float*)carve(GD * 4);

    const float invN = 1.0f / (float)GN;
    const int packTotal = GL * GD * GD;
    gin_pack_w<<<(packTotal + 255) / 256, 256, 0, stream>>>(W1, Bp1, packTotal);
    gin_pack_w<<<(packTotal + 255) / 256, 256, 0, stream>>>(W2, Bp2, packTotal);

    const int ndTot    = (int)ND;
    const int edBlks   = (GE + 7) / 8;
    const int gemmBlks = (GN + 127) / 128;
    const int statBlks = (GN + 255) / 256;

    gin_copy2<<<(ndTot + 255) / 256, 256, 0, stream>>>(h_in, h_buf, r_buf, ndTot);

    for (int i = 0; i < GL; ++i) {
        const int*      srcL = srcA + (size_t)i * GE;
        const int*      dstL = dstA + (size_t)i * GE;
        const float*    ewL  = ewA + (size_t)i * GE;
        const _Float16* B1L  = Bp1 + (size_t)i * GD * GD;
        const _Float16* B2L  = Bp2 + (size_t)i * GD * GD;

        // r = h + sum_e w*h[src]
        gin_scatter<<<edBlks, 256, 0, stream>>>(h_buf, srcL, dstL, ewL, r_buf, GE);

        // x = r @ W1 + b1
        gin_gemm_wmma<0><<<gemmBlks, 256, 0, stream>>>(r_buf, B1L, b1 + i * GD,
                                                       sA, tA, x_buf, GN);
        // BN1 stats/params
        gin_zero<<<1, 256, 0, stream>>>(stats, 2 * GD);
        gin_colstats<<<statBlks, GD, 0, stream>>>(x_buf, stats, GN);
        gin_bnparams<<<1, GD, 0, stream>>>(stats, bn1_g + i * GD, bn1_b + i * GD, sA, tA, invN);

        // x2 = relu(BN1(x)) @ W2 + b2   (BN1+ReLU fused into A load)
        gin_gemm_wmma<1><<<gemmBlks, 256, 0, stream>>>(x_buf, B2L, b2 + i * GD,
                                                       sA, tA, r_buf, GN);
        // BN_a stats/params
        gin_zero<<<1, 256, 0, stream>>>(stats, 2 * GD);
        gin_colstats<<<statBlks, GD, 0, stream>>>(r_buf, stats, GN);
        gin_bnparams<<<1, GD, 0, stream>>>(stats, bna_g + i * GD, bna_b + i * GD, sB, tB, invN);

        // y = relu(BN_a(x2)) in place + stats for BN_o
        gin_zero<<<1, 256, 0, stream>>>(stats, 2 * GD);
        gin_apply_stats<<<statBlks, GD, 0, stream>>>(r_buf, sB, tB, stats, GN);
        gin_bnparams<<<1, GD, 0, stream>>>(stats, bno_g + i * GD, bno_b + i * GD, sC, tC, invN);

        // h = relu(BN_o(y)); also re-init r for next layer
        float* hNext = (i == GL - 1) ? (float*)d_out : h_buf;
        gin_final<<<(ndTot + 255) / 256, 256, 0, stream>>>(r_buf, sC, tC, hNext, r_buf, ndTot);
    }
}
